// RMAC_5763846111441
// MI455X (gfx1250) — matrix operand order repaired and checked by reading, not compiled, not run on hardware
//
#include <hip/hip_runtime.h>
#include <hip/hip_bf16.h>
#include <stdint.h>

typedef __attribute__((ext_vector_type(16))) _Float16 v16h;
typedef __attribute__((ext_vector_type(8)))  float    v8f;
typedef __attribute__((ext_vector_type(4)))  float    v4f;

// ---------------------------------------------------------------------------
// Geometry for H=W=32, L=3 (matches _rmac_regions):
//   full 32x32 (counted twice: global + l=1), 4x 21x21 at {0,11}^2,
//   9x 16x16 at {0,8,16}^2.
// Distinct column/row ranges (same set for rows and cols):
//   0:[0,32) 1:[0,21) 2:[11,32) 3:[0,16) 4:[8,24) 5:[16,32)
// ---------------------------------------------------------------------------

// Phase 1: one wave32 per (b,c) 32x32 map. Async-copy map to LDS, compute all
// 14 unique region maxes in a single pass, write M[b][r][c] (r-major, 16 rows
// allocated, 14 used).
__global__ __launch_bounds__(256) void rmac_phase1(const float* __restrict__ x,
                                                   float* __restrict__ M)
{
    // 8 waves/block; per wave: 32 rows x 36 floats (144B rows -> 2-way max LDS
    // bank conflict instead of 16-way at 128B stride).
    __shared__ __align__(16) float rows[8 * 32 * 36];
    __shared__ float cmx[8][6][32];

    const int tid  = threadIdx.x;
    const int wave = tid >> 5;
    const int lane = tid & 31;
    const int map  = blockIdx.x * 8 + wave;   // flat (b,c), 65536 total
    const int b    = map >> 11;               // / 2048
    const int c    = map & 2047;

    // ---- async copy: 4KB map -> padded LDS (8 x b128 per lane-slot) --------
    const float* src = x + (size_t)map * 1024;
    const uint32_t ldsWaveBase = (uint32_t)(uintptr_t)&rows[wave * 1152];
#pragma unroll
    for (int t = 0; t < 8; ++t) {
        const int q      = t * 32 + lane;     // 16B chunk id 0..255
        const int row    = q >> 3;
        const int within = q & 7;
        const uint32_t ldst  = ldsWaveBase + (uint32_t)(row * 144 + within * 16);
        const uint64_t gaddr = (uint64_t)(uintptr_t)(src + q * 4);
        // Stream x non-temporally: 256MB read once, don't thrash L2 (M lives there).
        asm volatile("global_load_async_to_lds_b128 %0, %1, off th:TH_LOAD_NT"
                     :: "v"(ldst), "v"(gaddr) : "memory");
    }
    asm volatile("s_wait_asynccnt 0x0" ::: "memory");
    __syncthreads();

    // ---- lane == row: 6 column-range maxes from 8 chunk maxes --------------
    const float* rp = &rows[wave * 1152 + lane * 36];
    float chm[8], e11 = 0.f, e20 = 0.f;
#pragma unroll
    for (int k = 0; k < 8; ++k) {
        v4f q = *(const v4f*)(rp + k * 4);
        chm[k] = fmaxf(fmaxf(q[0], q[1]), fmaxf(q[2], q[3]));
        if (k == 2) e11 = q[3];   // element 11
        if (k == 5) e20 = q[0];   // element 20
    }
    const float m03 = fmaxf(fmaxf(chm[0], chm[1]), fmaxf(chm[2], chm[3])); // [0,16)
    const float m47 = fmaxf(fmaxf(chm[4], chm[5]), fmaxf(chm[6], chm[7])); // [16,32)
    const float m25 = fmaxf(fmaxf(chm[2], chm[3]), fmaxf(chm[4], chm[5])); // [8,24)
    cmx[wave][0][lane] = fmaxf(m03, m47);                 // [0,32)
    cmx[wave][1][lane] = fmaxf(fmaxf(m03, chm[4]), e20);  // [0,21)
    cmx[wave][2][lane] = fmaxf(e11, fmaxf(chm[3], m47));  // [11,32)
    cmx[wave][3][lane] = m03;                             // [0,16)
    cmx[wave][4][lane] = m25;                             // [8,24)
    cmx[wave][5][lane] = m47;                             // [16,32)
    __syncthreads();

    // ---- lanes 0..13: reduce over row range, emit region max ---------------
    if (lane < 14) {
        const int rs[14] = {0, 0, 0, 11, 11, 0, 0, 0, 8, 8, 8, 16, 16, 16};
        const int rl[14] = {32, 21, 21, 21, 21, 16, 16, 16, 16, 16, 16, 16, 16, 16};
        const int ci[14] = {0, 1, 2, 1, 2, 3, 4, 5, 3, 4, 5, 3, 4, 5};
        const int a = rs[lane], len = rl[lane];
        const float* cp = &cmx[wave][ci[lane]][0];
        float m = cp[a];
        for (int t = 1; t < len; ++t) m = fmaxf(m, cp[a + t]);
        M[((size_t)b * 16 + lane) * 2048 + c] = m;
    }
}

// Phase 2: per batch, sum-of-squares over C via diag(M * M^T) on the WMMA pipe
// (wave 0), then weighted combine out[b][c] = sum_r M[b][r][c] * inv[b][r]
// (all 256 threads). All traffic is L2-resident (3.7 MB).
__global__ __launch_bounds__(256) void rmac_phase2(const float* __restrict__ M,
                                                   float* __restrict__ out)
{
    __shared__ float inv[16];
    const int b   = blockIdx.x;
    const int tid = threadIdx.x;
    const float* Mb = M + (size_t)b * 16 * 2048;

    if (tid < 32) {               // wave 0, EXEC all-ones inside this branch
        const int lane = tid;
        const int m    = lane & 15;   // matrix row owned by this lane (A and B)
        const int half = lane >> 4;
        const float* rowp = Mb + (size_t)m * 2048;

        v8f acc = {};
        for (int cb = 0; cb < 2048; cb += 32) {
            v16h av, bv;
            // A (16x32 f16): lane L, M=L%16; element e -> K=16*half+e
#pragma unroll
            for (int i = 0; i < 16; ++i)
                av[i] = (_Float16)rowp[cb + 16 * half + i];
            // B = A^T (32x16 f16): lanes 0-15 K=0..15, lanes 16-31 K=16..31; N=L%16
#pragma unroll
            for (int i = 0; i < 16; ++i)
                bv[i] = (_Float16)rowp[cb + 16 * half + i];
            acc = __builtin_amdgcn_wmma_f32_16x16x32_f16(
                false, av, false, bv, (short)0, acc, false, false);
        }
        // Diagonal of C/D layout: row p at lane p (VGPR p), row p+8 at lane p+24.
        float ssq = 0.f;
        int r = -1;
#pragma unroll
        for (int p = 0; p < 8; ++p) {
            if (lane == p)      { ssq = acc[p]; r = p; }
            if (lane == 24 + p) { ssq = acc[p]; r = p + 8; }
        }
        if (r >= 0 && r < 14) {
            const float w = (r == 0) ? 2.0f : 1.0f;   // full-image region counted twice
            inv[r] = w / (sqrtf(ssq) + 1e-6f);
        }
    }
    __syncthreads();

    for (int c = tid; c < 2048; c += 256) {
        float s = 0.f;
#pragma unroll
        for (int r = 0; r < 14; ++r)
            s += Mb[(size_t)r * 2048 + c] * inv[r];
        out[(size_t)b * 2048 + c] = s;
    }
}

extern "C" void kernel_launch(void* const* d_in, const int* in_sizes, int n_in,
                              void* d_out, int out_size, void* d_ws, size_t ws_size,
                              hipStream_t stream)
{
    const float* x = (const float*)d_in[0];
    float* M   = (float*)d_ws;     // [B][16][2048] f32, 4 MB
    float* out = (float*)d_out;    // [B][2048] f32

    const int maps = in_sizes[0] / 1024;   // B*C = 65536 for the given shapes
    const int B    = maps / 2048;

    rmac_phase1<<<maps / 8, 256, 0, stream>>>(x, M);
    rmac_phase2<<<B, 256, 0, stream>>>(M, out);
}
